// DoubleConv_13546326851838
// MI455X (gfx1250) — compile-verified, matrix operand order checked by reading, not run on hardware
//
#include <hip/hip_runtime.h>

typedef float v8f   __attribute__((ext_vector_type(8)));
typedef __bf16 bf16x8  __attribute__((ext_vector_type(8)));
typedef __bf16 bf16x16 __attribute__((ext_vector_type(16)));

#define HYPER_OUT 294912   // 64*64*8*9
#define WGEN_PER_LB (9*8*128*128)   // per (layer,batch): [t][r][cout][cin]

// ---------------------------------------------------------------------------
// Kernel 1: tiny per-block MLPs  ->  E[16][128]   (m = l*8 + b*4 + n)
// ---------------------------------------------------------------------------
__global__ void mlp_kernel(const float* __restrict__ seidel,
                           const float* __restrict__ m1w1, const float* __restrict__ m1b1,
                           const float* __restrict__ m1w2, const float* __restrict__ m1b2,
                           const float* __restrict__ m2w1, const float* __restrict__ m2b1,
                           const float* __restrict__ m2w2, const float* __restrict__ m2b2,
                           float* __restrict__ E)
{
    int m = blockIdx.x;            // 0..15
    int l = m >> 3, b = (m >> 2) & 1, n = m & 3, h = threadIdx.x;
    const float* w1 = (l ? m2w1 : m1w1) + n * 6 * 128;
    const float* b1 = (l ? m2b1 : m1b1) + n * 128;
    const float* w2 = (l ? m2w2 : m1w2) + n * 128 * 128;
    const float* b2 = (l ? m2b2 : m1b2) + n * 128;

    __shared__ float e1[128];
    float acc = b1[h];
#pragma unroll
    for (int i = 0; i < 6; ++i) acc += seidel[b * 6 + i] * w1[i * 128 + h];
    e1[h] = fmaxf(acc, 0.f);
    __syncthreads();
    float a2 = b2[h];
    for (int j = 0; j < 128; ++j) a2 += e1[j] * w2[j * 128 + h];
    E[m * 128 + h] = fmaxf(a2, 0.f);
}

// ---------------------------------------------------------------------------
// Kernel 2: hypernet GEMV  Wrow[m][c] = E[m] . hyper_w[:,c] + hyper_b[c]
// memory-bound on hyper_w (151 MB, read once); scatter into conv layout:
// Wgen[(l*2+b)][t][r][cout][cin]   (so the conv reads contiguous 64KB slabs)
// ---------------------------------------------------------------------------
__global__ void hyper_kernel(const float* __restrict__ E,
                             const float* __restrict__ HW,
                             const float* __restrict__ HB,
                             float* __restrict__ Wgen)
{
    __shared__ float Es[16][128];
    int tid = threadIdx.x;
#pragma unroll
    for (int r = 0; r < 8; ++r) { int e = r * 256 + tid; Es[e >> 7][e & 127] = E[e]; }
    __syncthreads();

    int c = blockIdx.x * 256 + tid;      // 0..294911 (1152 blocks)
    float acc[16];
#pragma unroll
    for (int m = 0; m < 16; ++m) acc[m] = 0.f;

    for (int k = 0; k < 128; ++k) {
        float hw = HW[(size_t)k * HYPER_OUT + c];
#pragma unroll
        for (int m = 0; m < 16; ++m) acc[m] += Es[m][k] * hw;
    }
    float bias = HB[c];
    // c = ((o64*64 + i64)*8 + rr)*9 + t
    int t   = c % 9;
    int c9  = c / 9;
    int rr  = c9 & 7;
    int c72 = c9 >> 3;
    int i64 = c72 & 63;
    int o64 = c72 >> 6;
#pragma unroll
    for (int m = 0; m < 16; ++m) {
        int n  = m & 3;
        int lb = m >> 2;                       // (l*2 + b)
        int cout = (n >> 1) * 64 + o64;
        int cin  = (n & 1) * 64 + i64;
        Wgen[(size_t)lb * WGEN_PER_LB + (size_t)(t * 8 + rr) * 16384 + cout * 128 + cin]
            = acc[m] + bias;
    }
}

// ---------------------------------------------------------------------------
// Kernel 3: transpose x[b][c][h][w] -> xT[b][w][c][h]  (coalesced both sides)
// ---------------------------------------------------------------------------
__global__ void transpose_kernel(const float* __restrict__ x, float* __restrict__ xT)
{
    __shared__ float tile[32][33];
    int bc = blockIdx.z;                 // b*128 + c
    int h0 = blockIdx.y * 32, w0 = blockIdx.x * 32;
    int tx = threadIdx.x, ty = threadIdx.y;      // 32 x 8
    const float* src = x + ((size_t)bc * 256 + h0) * 256 + w0;
#pragma unroll
    for (int i = 0; i < 32; i += 8) tile[ty + i][tx] = src[(size_t)(ty + i) * 256 + tx];
    __syncthreads();
    int b = bc >> 7, c = bc & 127;
    float* dst = xT + (((size_t)b * 256 + w0) * 128 + c) * 256 + h0;   // + w*32768 + h
#pragma unroll
    for (int i = 0; i < 32; i += 8) dst[(size_t)(ty + i) * 32768 + tx] = tile[tx][ty + i];
}

// ---------------------------------------------------------------------------
// Kernel 4: WMMA LRI conv. One workgroup = one column w, 128 cout x 64 h.
// GEMM: out[cout,h] = sum_{t=(ki,kj)} sum_cin Wt[cout][cin] * Xt[h][cin]
// Wt interpolated on the fly between radii slabs i0/i1 (L2-resident).
// ---------------------------------------------------------------------------
__device__ inline bf16x16 ldfrag(const __bf16* p0, const __bf16* p1)
{
    bf16x8 a = *(const bf16x8*)p0;
    bf16x8 b = *(const bf16x8*)p1;
    return __builtin_shufflevector(a, b, 0,1,2,3,4,5,6,7,8,9,10,11,12,13,14,15);
}

__global__ __launch_bounds__(256) void lri_conv_kernel(
    const float* __restrict__ srcT,    // [b][w][cin][h]
    const float* __restrict__ WgenL,   // layer base: [b][t][r][cout][cin]
    const float* __restrict__ ssIn,    // per-(b,cin) scale/shift (phase 2) or null
    float* __restrict__ dst, int phase)
{
    const int w   = blockIdx.x;        // 0..255
    const int h0  = blockIdx.y * 64;   // 0..192
    const int b   = blockIdx.z;        // 0..1
    const int tid = threadIdx.x;
    const int lane = tid & 31, wv = tid >> 5;

    __shared__ __align__(16) __bf16 Wt[128][136];   // [cout][cin] padded (+8)
    __shared__ __align__(16) __bf16 Xt[64][136];    // [h][cin]    padded (+8)

    // trilinear interp position along radii (align_corners=False)
    float pos = (w + 0.5f) * (8.0f / 256.0f) - 0.5f;
    pos = fminf(fmaxf(pos, 0.0f), 7.0f);
    int   i0 = (int)floorf(pos);
    int   i1 = min(i0 + 1, 7);
    float f  = pos - (float)i0;

    v8f acc[4];
#pragma unroll
    for (int n = 0; n < 4; ++n)
#pragma unroll
        for (int i = 0; i < 8; ++i) acc[n][i] = 0.f;

    const int koff = (lane < 16) ? 0 : 8;      // ISA 16-bit A/B matrix K layout
    const int mrow = wv * 16 + (lane & 15);

    for (int t = 0; t < 9; ++t) {
        int ki = t / 3, kj = t % 3;
        __syncthreads();

        // ---- stage interpolated weight slab (16384 f32 -> bf16) ----
        const float* w0p = WgenL + ((size_t)(b * 9 + t) * 8 + i0) * 16384;
        const float* w1p = WgenL + ((size_t)(b * 9 + t) * 8 + i1) * 16384;
        for (int e = tid; e < 16384; e += 256) {
            float v0 = w0p[e];
            float v  = v0 + f * (w1p[e] - v0);
            Wt[e >> 7][e & 127] = (__bf16)v;
        }

        // ---- stage x patch: Xt[h][cin] = src[cin][(h0+h+ki-1)%256][w+kj-1] ----
        int wc = w + kj - 1;
        if (wc >= 0 && wc < 256) {
            const float* xp = srcT + ((size_t)b * 256 + wc) * 128 * 256;
            for (int e = tid; e < 8192; e += 256) {
                int hl = e & 63, ci = e >> 6;
                int hh = (h0 + hl + ki - 1 + 256) & 255;   // circular in H
                float v = xp[(size_t)ci * 256 + hh];
                if (phase == 2) {
                    float sc = ssIn[(b * 128 + ci) * 2 + 0];
                    float sh = ssIn[(b * 128 + ci) * 2 + 1];
                    v = fmaxf(v * sc + sh, 0.f);            // BN1 + ReLU on load
                }
                Xt[hl][ci] = (__bf16)v;
            }
        } else {
            for (int e = tid; e < 8192; e += 256) Xt[e & 63][e >> 6] = (__bf16)0.f;
        }
        __syncthreads();

        // ---- WMMA: 4 K-chunks of 32 cin, 4 h-tiles per wave ----
        bf16x16 afrag[4];
#pragma unroll
        for (int kc = 0; kc < 4; ++kc) {
            int kb = kc * 32 + koff;
            afrag[kc] = ldfrag(&Wt[mrow][kb], &Wt[mrow][kb + 16]);
        }
#pragma unroll
        for (int n = 0; n < 4; ++n) {
            int nrow = n * 16 + (lane & 15);
#pragma unroll
            for (int kc = 0; kc < 4; ++kc) {
                int kb = kc * 32 + koff;
                bf16x16 bfrag = ldfrag(&Xt[nrow][kb], &Xt[nrow][kb + 16]);
                acc[n] = __builtin_amdgcn_wmma_f32_16x16x32_bf16(
                    false, afrag[kc], false, bfrag, (short)0, acc[n], false, false);
            }
        }
    }

    // ---- write out (D layout: VGPR r -> M=r(+8 for lanes>=16), N=lane%16) ----
#pragma unroll
    for (int n = 0; n < 4; ++n) {
        int h = h0 + n * 16 + (lane & 15);
#pragma unroll
        for (int r = 0; r < 8; ++r) {
            int cout = wv * 16 + r + ((lane >> 4) << 3);
            float v = acc[n][r];
            if (phase == 1)
                dst[(((size_t)b * 256 + w) * 128 + cout) * 256 + h] = v;   // y1T layout
            else
                dst[(((size_t)b * 128 + cout) * 256 + h) * 256 + w] = v;   // BCHW
        }
    }
}

// ---------------------------------------------------------------------------
// Kernel 5: per-(b,channel) BN stats over 65536 pixels -> scale/shift
// layout 0: y in [b][w][c][h] (y1T);  layout 1: y in [b][c][h][w]
// ---------------------------------------------------------------------------
__global__ void stats_kernel(const float* __restrict__ y,
                             const float* __restrict__ gamma,
                             const float* __restrict__ beta,
                             float* __restrict__ ss, int layout)
{
    int b = blockIdx.x >> 7, c = blockIdx.x & 127, tid = threadIdx.x;
    float s = 0.f, s2 = 0.f;
    if (layout == 0) {
        for (int w = 0; w < 256; ++w) {
            float v = y[(((size_t)b * 256 + w) * 128 + c) * 256 + tid];
            s += v; s2 += v * v;
        }
    } else {
        const float* p = y + ((size_t)b * 128 + c) * 65536;
        for (int i = 0; i < 256; ++i) {
            float v = p[i * 256 + tid];
            s += v; s2 += v * v;
        }
    }
    __shared__ float rs[256], rq[256];
    rs[tid] = s; rq[tid] = s2;
    __syncthreads();
    for (int off = 128; off > 0; off >>= 1) {
        if (tid < off) { rs[tid] += rs[tid + off]; rq[tid] += rq[tid + off]; }
        __syncthreads();
    }
    if (tid == 0) {
        float mean = rs[0] * (1.f / 65536.f);
        float var  = rq[0] * (1.f / 65536.f) - mean * mean;
        float sc   = gamma[c] * rsqrtf(var + 1e-5f);
        ss[(b * 128 + c) * 2 + 0] = sc;
        ss[(b * 128 + c) * 2 + 1] = beta[c] - mean * sc;
    }
}

// ---------------------------------------------------------------------------
// Kernel 6: in-place BN2 + ReLU on d_out (BCHW)
// ---------------------------------------------------------------------------
__global__ void apply_bn_kernel(float* __restrict__ y, const float* __restrict__ ss)
{
    size_t idx = (size_t)blockIdx.x * 256 + threadIdx.x;   // 16,777,216 total
    int c = (int)((idx >> 16) & 127);
    int b = (int)(idx >> 23);
    float sc = ss[(b * 128 + c) * 2 + 0];
    float sh = ss[(b * 128 + c) * 2 + 1];
    y[idx] = fmaxf(y[idx] * sc + sh, 0.f);
}

// ---------------------------------------------------------------------------
extern "C" void kernel_launch(void* const* d_in, const int* in_sizes, int n_in,
                              void* d_out, int out_size, void* d_ws, size_t ws_size,
                              hipStream_t stream)
{
    const float* x      = (const float*)d_in[0];
    const float* seidel = (const float*)d_in[1];
    const float* m1w1   = (const float*)d_in[2];
    const float* m1b1   = (const float*)d_in[3];
    const float* m1w2   = (const float*)d_in[4];
    const float* m1b2   = (const float*)d_in[5];
    const float* m2w1   = (const float*)d_in[6];
    const float* m2b1   = (const float*)d_in[7];
    const float* m2w2   = (const float*)d_in[8];
    const float* m2b2   = (const float*)d_in[9];
    const float* hyperw = (const float*)d_in[10];
    const float* hyperb = (const float*)d_in[11];
    const float* g1     = (const float*)d_in[12];
    const float* b1     = (const float*)d_in[13];
    const float* g2     = (const float*)d_in[14];
    const float* b2     = (const float*)d_in[15];
    float* out = (float*)d_out;

    char* ws = (char*)d_ws;
    float* E    = (float*)(ws);                                   //   8 KB
    float* ss1  = (float*)(ws + 8192);                            //   2 KB
    float* ss2  = (float*)(ws + 10240);                           //   2 KB
    float* Wgen = (float*)(ws + 12288);                           // 18.87 MB
    float* xT   = (float*)(ws + 12288 + 18874368);                // 67.1 MB
    float* y1T  = (float*)(ws + 12288 + 18874368 + 67108864);     // 67.1 MB

    // 1) hypernet MLPs
    mlp_kernel<<<16, 128, 0, stream>>>(seidel, m1w1, m1b1, m1w2, m1b2,
                                       m2w1, m2b1, m2w2, m2b2, E);
    // 2) hyper GEMV + scatter into conv layout
    hyper_kernel<<<HYPER_OUT / 256, 256, 0, stream>>>(E, hyperw, hyperb, Wgen);
    // 3) transpose activations to [b][w][c][h]
    transpose_kernel<<<dim3(8, 8, 256), dim3(32, 8), 0, stream>>>(x, xT);
    // 4) conv1 (WMMA) -> y1T (raw)
    lri_conv_kernel<<<dim3(256, 4, 2), 256, 0, stream>>>(xT, Wgen, nullptr, y1T, 1);
    // 5) BN1 stats
    stats_kernel<<<256, 256, 0, stream>>>(y1T, g1, b1, ss1, 0);
    // 6) conv2 (WMMA, BN1+ReLU on load) -> d_out (raw, BCHW)
    lri_conv_kernel<<<dim3(256, 4, 2), 256, 0, stream>>>(
        y1T, Wgen + (size_t)2 * WGEN_PER_LB, ss1, out, 2);
    // 7) BN2 stats
    stats_kernel<<<256, 256, 0, stream>>>(out, g2, b2, ss2, 1);
    // 8) BN2 + ReLU in place
    apply_bn_kernel<<<65536, 256, 0, stream>>>(out, ss2);
}
